// InvResMLP_10101763080205
// MI455X (gfx1250) — matrix operand order repaired and checked by reading, not compiled, not run on hardware
//
#include <hip/hip_runtime.h>
#include <hip/hip_bf16.h>

// Problem constants (from reference): B=8, N=4096, C=64, K=32
#define PB 8
#define PN 4096
#define PC 64
#define PK 32
#define PM (PB * PN)          // 32768 rows for all GEMMs
#define RADIUS2 0.01f
#define BN_EPS 1e-5f

typedef __attribute__((ext_vector_type(16))) _Float16 v16h;
typedef __attribute__((ext_vector_type(8)))  float    v8f;

// ---------------------------------------------------------------------------
// WMMA fragment loaders (CDNA5 wave32 layouts, CDNA5 ISA §7.12.2)
// ---------------------------------------------------------------------------

// A fragment: 16x32 f16 tile of row-major activations A[M][64].
// lane L holds row (m0 + L%16); half i -> K = kbase + (i/8)*16 + (L>=16?8:0) + i%8
__device__ __forceinline__ v16h load_a_frag(const float* __restrict__ A,
                                            int m0, int kbase, int lane) {
    v16h a;
    const float* row = A + (size_t)(m0 + (lane & 15)) * PC;
    const int koff = kbase + ((lane >> 4) << 3);
#pragma unroll
    for (int h = 0; h < 2; ++h)
#pragma unroll
        for (int j = 0; j < 8; ++j)
            a[h * 8 + j] = (_Float16)row[koff + h * 16 + j];
    return a;
}

// B fragment: 32x16 f16 tile with B[k][n] = W[n][k] (y = x @ W^T).
// lane L holds column (n0 + L%16); half i -> K = kbase + (i/8)*16 + (L>=16?8:0) + i%8
__device__ __forceinline__ v16h load_b_frag(const float* __restrict__ W,
                                            int n0, int kbase, int lane) {
    v16h b;
    const float* wrow = W + (size_t)(n0 + (lane & 15)) * PC + kbase + ((lane >> 4) << 3);
#pragma unroll
    for (int i = 0; i < 16; ++i) b[i] = (_Float16)wrow[(i >> 3) * 16 + (i & 7)];
    return b;
}

// ---------------------------------------------------------------------------
// GEMM (M=32768, N=64, K=64) via v_wmma_f32_16x16x32_f16, fused BN statistics.
// blockDim = 128 (4 waves); each wave computes a 16x64 output slab.
// ---------------------------------------------------------------------------
__global__ void gemm_bn_stats_kernel(const float* __restrict__ A,
                                     const float* __restrict__ W,
                                     float* __restrict__ Y,
                                     float* __restrict__ gsum,
                                     float* __restrict__ gsq) {
    __shared__ float s_sum[PC];
    __shared__ float s_sq[PC];
    const int tid = threadIdx.x;
    if (tid < PC) { s_sum[tid] = 0.f; s_sq[tid] = 0.f; }
    __syncthreads();

    const int lane = tid & 31;
    const int wave = tid >> 5;
    const int m0 = (blockIdx.x * (blockDim.x >> 5) + wave) * 16;

    const v16h a0 = load_a_frag(A, m0, 0, lane);
    const v16h a1 = load_a_frag(A, m0, 32, lane);

#pragma unroll
    for (int t = 0; t < 4; ++t) {
        const int n0 = t * 16;
        const v16h b0 = load_b_frag(W, n0, 0, lane);
        const v16h b1 = load_b_frag(W, n0, 32, lane);
        v8f c = {};
        // D = A(K=0..31)  x B + 0
        c = __builtin_amdgcn_wmma_f32_16x16x32_f16(false, a0, false, b0,
                                                   (short)0, c, false, false);
        // D += A(K=32..63) x B
        c = __builtin_amdgcn_wmma_f32_16x16x32_f16(false, a1, false, b1,
                                                   (short)0, c, false, false);

        // C/D layout: lane holds col n0+(lane%16), rows mbase..mbase+7
        const int n = lane & 15;
        const int mbase = m0 + ((lane >> 4) << 3);
        float lsum = 0.f, lsq = 0.f;
#pragma unroll
        for (int r = 0; r < 8; ++r) {
            const float v = c[r];
            Y[(size_t)(mbase + r) * PC + n0 + n] = v;
            lsum += v;
            lsq += v * v;
        }
        // lanes j and j+16 hold the same channel -> pairwise combine
        lsum += __shfl_xor(lsum, 16, 32);
        lsq  += __shfl_xor(lsq, 16, 32);
        if (lane < 16) {
            atomicAdd(&s_sum[n0 + n], lsum);   // ds_add_f32
            atomicAdd(&s_sq[n0 + n], lsq);
        }
    }
    __syncthreads();
    if (tid < PC) {
        atomicAdd(&gsum[tid], s_sum[tid]);     // one global atomic per block-channel
        atomicAdd(&gsq[tid], s_sq[tid]);
    }
}

// ---------------------------------------------------------------------------
// BN finalize: 64 channels -> per-channel scale/shift
// ---------------------------------------------------------------------------
__global__ void bn_finalize_kernel(const float* __restrict__ gsum,
                                   const float* __restrict__ gsq,
                                   const float* __restrict__ gamma,
                                   const float* __restrict__ beta,
                                   float* __restrict__ scale,
                                   float* __restrict__ shift,
                                   float inv_count) {
    const int c = threadIdx.x;
    if (c < PC) {
        const float mean = gsum[c] * inv_count;
        const float var = gsq[c] * inv_count - mean * mean;
        const float a = gamma[c] * rsqrtf(var + BN_EPS);
        scale[c] = a;
        shift[c] = beta[c] - mean * a;
    }
}

__global__ void bn_apply_relu_kernel(const float* __restrict__ Y,
                                     const float* __restrict__ scale,
                                     const float* __restrict__ shift,
                                     float* __restrict__ X, int total) {
    const int i = blockIdx.x * blockDim.x + threadIdx.x;
    if (i < total) {
        const int c = i & (PC - 1);
        X[i] = fmaxf(Y[i] * scale[c] + shift[c], 0.f);
    }
}

// ---------------------------------------------------------------------------
// Ball query: for each point, first K=32 indices (ascending) with d2 < r^2.
// LDS-tiled O(N^2) scan; ascending scan == sort-and-truncate of the reference.
// Self is always in range (d2=0), so "first" is always valid for padding.
// ---------------------------------------------------------------------------
__global__ void ball_query_kernel(const float* __restrict__ pos,
                                  int* __restrict__ idx) {
    __shared__ float spx[256], spy[256], spz[256];
    const int b = blockIdx.y;
    const int n = blockIdx.x * blockDim.x + threadIdx.x;
    const float* pb = pos + (size_t)b * PN * 3;
    const float px = pb[(size_t)n * 3 + 0];
    const float py = pb[(size_t)n * 3 + 1];
    const float pz = pb[(size_t)n * 3 + 2];

    int cnt = 0, first = n;
    int* myidx = idx + ((size_t)b * PN + n) * PK;

    for (int tile = 0; tile < PN; tile += 256) {
        const int m = tile + threadIdx.x;
        spx[threadIdx.x] = pb[(size_t)m * 3 + 0];
        spy[threadIdx.x] = pb[(size_t)m * 3 + 1];
        spz[threadIdx.x] = pb[(size_t)m * 3 + 2];
        __syncthreads();
        if (cnt < PK) {
            for (int j = 0; j < 256 && cnt < PK; ++j) {
                const float dx = px - spx[j];
                const float dy = py - spy[j];
                const float dz = pz - spz[j];
                const float d2 = dx * dx + dy * dy + dz * dz;
                if (d2 < RADIUS2) {
                    if (cnt == 0) first = tile + j;
                    myidx[cnt++] = tile + j;
                }
            }
        }
        __syncthreads();
    }
    for (int k = cnt; k < PK; ++k) myidx[k] = first;
}

// ---------------------------------------------------------------------------
// Gather + max over K neighbors, fused with the pe pass-through copy so the
// 256 MB pe tensor crosses HBM exactly once each way. Non-temporal hints keep
// the stream from evicting x1/idx out of L2.
// g[b,n,c] = max_k ( x1[b, idx[b,n,k], c] + pe[b,c,k,n] )
// ---------------------------------------------------------------------------
__global__ void gather_max_kernel(const float* __restrict__ x1,
                                  const float* __restrict__ pe,
                                  const int* __restrict__ idx,
                                  float* __restrict__ g,
                                  float* __restrict__ pe_out) {
    const int b = blockIdx.y;
    const int n = blockIdx.x * blockDim.x + threadIdx.x;
    const float* x1b = x1 + (size_t)b * PN * PC;
    const int* myidx = idx + ((size_t)b * PN + n) * PK;

    int id[PK];
#pragma unroll
    for (int k = 0; k < PK; ++k) id[k] = myidx[k];

    const size_t peb = (size_t)b * PC * PK * PN;
    float* gout = g + ((size_t)b * PN + n) * PC;

    for (int c = 0; c < PC; ++c) {
        const size_t pec = peb + (size_t)c * PK * PN + n;
        if (c + 1 < PC)  // global_prefetch_b8 for the next channel's stream
            __builtin_prefetch(&pe[pec + (size_t)PK * PN], 0, 1);
        float acc = -INFINITY;
#pragma unroll
        for (int k = 0; k < PK; ++k) {
            const float p = __builtin_nontemporal_load(&pe[pec + (size_t)k * PN]);
            __builtin_nontemporal_store(p, &pe_out[pec + (size_t)k * PN]);
            acc = fmaxf(acc, x1b[(size_t)id[k] * PC + c] + p);
        }
        gout[c] = acc;
    }
}

// ---------------------------------------------------------------------------
// Small utilities
// ---------------------------------------------------------------------------
__global__ void zero_kernel(float* __restrict__ p, int total) {
    for (int i = blockIdx.x * blockDim.x + threadIdx.x; i < total;
         i += gridDim.x * blockDim.x)
        p[i] = 0.f;
}

__global__ void copy_kernel(const float* __restrict__ src, float* __restrict__ dst,
                            int total) {
    for (int i = blockIdx.x * blockDim.x + threadIdx.x; i < total;
         i += gridDim.x * blockDim.x)
        dst[i] = src[i];
}

// out = relu(x1 + BN(y))
__global__ void bn_residual_relu_kernel(const float* __restrict__ Y,
                                        const float* __restrict__ scale,
                                        const float* __restrict__ shift,
                                        const float* __restrict__ x1,
                                        float* __restrict__ out, int total) {
    const int i = blockIdx.x * blockDim.x + threadIdx.x;
    if (i < total) {
        const int c = i & (PC - 1);
        const float z = Y[i] * scale[c] + shift[c];
        out[i] = fmaxf(x1[i] + z, 0.f);
    }
}

// ---------------------------------------------------------------------------
// Launch
// ---------------------------------------------------------------------------
extern "C" void kernel_launch(void* const* d_in, const int* in_sizes, int n_in,
                              void* d_out, int out_size, void* d_ws, size_t ws_size,
                              hipStream_t stream) {
    (void)in_sizes; (void)n_in; (void)out_size; (void)ws_size;

    const float* pos = (const float*)d_in[0];
    const float* x   = (const float*)d_in[1];
    const float* pe  = (const float*)d_in[2];
    const float* w1  = (const float*)d_in[3];
    const float* g1  = (const float*)d_in[4];
    const float* b1  = (const float*)d_in[5];
    const float* w2a = (const float*)d_in[6];
    const float* g2a = (const float*)d_in[7];
    const float* b2a = (const float*)d_in[8];
    const float* w2b = (const float*)d_in[9];
    const float* g2b = (const float*)d_in[10];
    const float* b2b = (const float*)d_in[11];

    // Output: pos (B*N*3) | relu(x1+z) (B*N*C) | pe (B*C*K*N), flat fp32
    float* out_pos = (float*)d_out;
    float* out_x   = out_pos + (size_t)PB * PN * 3;
    float* out_pe  = out_x + (size_t)PM * PC;

    // Workspace carve-up (~36 MB)
    char* ws = (char*)d_ws;
    float* x1buf = (float*)ws; ws += sizeof(float) * (size_t)PM * PC;   // 8 MB
    float* ybuf  = (float*)ws; ws += sizeof(float) * (size_t)PM * PC;   // 8 MB
    float* gbuf  = (float*)ws; ws += sizeof(float) * (size_t)PM * PC;   // 8 MB
    float* zbuf  = (float*)ws; ws += sizeof(float) * (size_t)PM * PC;   // 8 MB
    int*   idx   = (int*)ws;   ws += sizeof(int)   * (size_t)PM * PK;   // 4 MB
    float* stats = (float*)ws;                                          // 3*256 floats
    // per stage s: sum=+0, sumsq=+64, scale=+128, shift=+192
    float* st0 = stats;
    float* st1 = stats + 256;
    float* st2 = stats + 512;

    const int total = PM * PC;                // 2,097,152
    const int eltBlocks = total / 256;        // 8192
    const float invM = 1.0f / (float)PM;
    const dim3 gemmGrid(PM / 16 / 4);         // 512 blocks x 4 waves
    const dim3 ptGrid(PN / 256, PB);          // 16 x 8

    // Stats buffers must be zero every call (ws is poisoned once, not reset).
    zero_kernel<<<3, 256, 0, stream>>>(stats, 768);

    // Stage 1: x1 = relu(BN(x @ w1^T))
    gemm_bn_stats_kernel<<<gemmGrid, 128, 0, stream>>>(x, w1, ybuf, st0, st0 + 64);
    bn_finalize_kernel<<<1, 64, 0, stream>>>(st0, st0 + 64, g1, b1,
                                             st0 + 128, st0 + 192, invM);
    bn_apply_relu_kernel<<<eltBlocks, 256, 0, stream>>>(ybuf, st0 + 128, st0 + 192,
                                                        x1buf, total);

    // Ball query + neighborhood max (fused pe pass-through), pos pass-through
    ball_query_kernel<<<ptGrid, 256, 0, stream>>>(pos, idx);
    gather_max_kernel<<<ptGrid, 256, 0, stream>>>(x1buf, pe, idx, gbuf, out_pe);
    copy_kernel<<<384, 256, 0, stream>>>(pos, out_pos, PB * PN * 3);

    // Stage 2a: z = relu(BN(g @ w2a^T))
    gemm_bn_stats_kernel<<<gemmGrid, 128, 0, stream>>>(gbuf, w2a, ybuf, st1, st1 + 64);
    bn_finalize_kernel<<<1, 64, 0, stream>>>(st1, st1 + 64, g2a, b2a,
                                             st1 + 128, st1 + 192, invM);
    bn_apply_relu_kernel<<<eltBlocks, 256, 0, stream>>>(ybuf, st1 + 128, st1 + 192,
                                                        zbuf, total);

    // Stage 2b: out = relu(x1 + BN(z @ w2b^T))
    gemm_bn_stats_kernel<<<gemmGrid, 128, 0, stream>>>(zbuf, w2b, ybuf, st2, st2 + 64);
    bn_finalize_kernel<<<1, 64, 0, stream>>>(st2, st2 + 64, g2b, b2b,
                                             st2 + 128, st2 + 192, invM);
    bn_residual_relu_kernel<<<eltBlocks, 256, 0, stream>>>(ybuf, st2 + 128, st2 + 192,
                                                           x1buf, out_x, total);
}
